// GCN_Net_30477087932866
// MI455X (gfx1250) — compile-verified, matrix operand order checked by reading, not run on hardware
//
#include <hip/hip_runtime.h>
#include <hip/hip_bf16.h>

// ---------------------------------------------------------------------------
// Types for CDNA5 WMMA (wave32): v_wmma_f32_16x16x32_bf16
// ---------------------------------------------------------------------------
typedef __bf16 v16bf __attribute__((ext_vector_type(16)));
typedef __bf16 v8bf  __attribute__((ext_vector_type(8)));
typedef __bf16 v4bf  __attribute__((ext_vector_type(4)));
typedef float  v8f   __attribute__((ext_vector_type(8)));

// Tensor Data Mover descriptor groups (probe-verified builtin arg types)
typedef unsigned int tdm_v4u __attribute__((ext_vector_type(4)));
typedef int          tdm_v8i __attribute__((ext_vector_type(8)));
typedef int          tdm_v4i __attribute__((ext_vector_type(4)));

#define N_NODES 65536
#define N_EDGES 524288
#define HID     128

// ---------------------------------------------------------------------------
// Degree / normalization kernels
// ---------------------------------------------------------------------------
__global__ void deg_init(float* __restrict__ deg) {
  int i = blockIdx.x * 256 + threadIdx.x;
  deg[i] = 1.0f;  // self loop
}

__global__ void deg_accum(const int* __restrict__ dst, float* __restrict__ deg) {
  int e = blockIdx.x * 256 + threadIdx.x;
  if (e < N_EDGES) unsafeAtomicAdd(&deg[dst[e]], 1.0f);
}

__global__ void dinv_finish(float* __restrict__ deg) {
  int i = blockIdx.x * 256 + threadIdx.x;
  deg[i] = rsqrtf(deg[i]);  // deg >= 1 always
}

// ---------------------------------------------------------------------------
// Shared WMMA inner loop: 8 waves x 16 rows, N in 16-wide tiles.
// As: [row][k] bf16, Bs: [n][k] bf16 (W transposed).
// ---------------------------------------------------------------------------
template<int N, bool BIAS, bool RELU, bool WRITE_BF16>
__device__ __forceinline__ void wmma_core(const __bf16* As, const __bf16* Bs,
                                          const float* bias, float* Y,
                                          __bf16* Yb, long row0) {
  const int tid  = threadIdx.x;
  const int wave = tid >> 5;
  const int lane = tid & 31;
  const int mloc = wave * 16;
  const int lm   = lane & 15;
  const int lkb  = (lane >> 4) << 3;  // 0 or 8 (16-bit A/B VGPR layout)

  // Preload the 4 A fragments (K = 0,32,64,96), two b128 LDS loads each.
  v16bf afrag[4];
#pragma unroll
  for (int kk = 0; kk < 4; ++kk) {
    const __bf16* ap = &As[(mloc + lm) * 128 + kk * 32 + lkb];
    union { v16bf v; v8bf h[2]; } u;
    u.h[0] = *(const v8bf*)(ap);       // K = kbase + 0..7
    u.h[1] = *(const v8bf*)(ap + 16);  // K = kbase + 16..23
    afrag[kk] = u.v;
  }

#pragma unroll
  for (int nt = 0; nt < N / 16; ++nt) {
    v8f acc = {};
    const int ncol = nt * 16 + lm;
#pragma unroll
    for (int kk = 0; kk < 4; ++kk) {
      const __bf16* bp = &Bs[ncol * 128 + kk * 32 + lkb];
      union { v16bf v; v8bf h[2]; } u;
      u.h[0] = *(const v8bf*)(bp);
      u.h[1] = *(const v8bf*)(bp + 16);
      acc = __builtin_amdgcn_wmma_f32_16x16x32_bf16(
          false, afrag[kk], false, u.v, (short)0, acc, false, false);
    }
    // C/D layout: VGPR v, lanes 0-15 -> M=v, lanes 16-31 -> M=8+v; col = lane&15
    const int rbase = mloc + ((lane >> 4) << 3);
    float bv = 0.0f;
    if (BIAS) bv = bias[ncol];
#pragma unroll
    for (int v = 0; v < 8; ++v) {
      float val = acc[v] + bv;
      if (RELU) val = fmaxf(val, 0.0f);
      Y[(row0 + rbase + v) * N + ncol] = val;
      if (WRITE_BF16) Yb[(row0 + rbase + v) * N + ncol] = (__bf16)val;
    }
  }
}

// ---------------------------------------------------------------------------
// Stage W (row-major [k][n] fp32) transposed into Bs[n][k] as bf16.
// ---------------------------------------------------------------------------
template<int N>
__device__ __forceinline__ void stage_W(const float* __restrict__ W, __bf16* Bs) {
  for (int idx = threadIdx.x; idx < 128 * N; idx += 256) {
    const int k = idx / N;
    const int n = idx - k * N;
    Bs[n * 128 + k] = (__bf16)W[idx];
  }
}

// ---------------------------------------------------------------------------
// GEMM with fp32 A input (converted in flight): Y = X @ W (+bias)(relu)
// ---------------------------------------------------------------------------
template<int N, bool BIAS, bool RELU, bool WRITE_BF16>
__global__ void __launch_bounds__(256)
gemm_wmma_f32(const float* __restrict__ X, const float* __restrict__ W,
              const float* __restrict__ bias, float* __restrict__ Y,
              __bf16* __restrict__ Yb) {
  __shared__ __align__(16) __bf16 As[128 * 128];
  __shared__ __align__(16) __bf16 Bs[N * 128];

  stage_W<N>(W, Bs);

  // Stage 128-row fp32 X tile as bf16: float4 loads, 8-byte packed stores.
  const long row0 = (long)blockIdx.x * 128;
  const float4* X4 = (const float4*)(X + row0 * 128);
  for (int i = threadIdx.x; i < (128 * 128) / 4; i += 256) {
    const float4 v = X4[i];
    *(v4bf*)(&As[i * 4]) =
        (v4bf){(__bf16)v.x, (__bf16)v.y, (__bf16)v.z, (__bf16)v.w};
  }
  __syncthreads();

  wmma_core<N, BIAS, RELU, WRITE_BF16>(As, Bs, bias, Y, Yb, row0);
}

// ---------------------------------------------------------------------------
// GEMM with bf16 A input staged by the Tensor Data Mover:
// a 128-row bf16 tile is one contiguous 32KB block -> single TDM descriptor,
// issued by wave 0, completion via s_wait_tensorcnt. (CDNA5 TDM, ISA ch.7/8)
// ---------------------------------------------------------------------------
__global__ void __launch_bounds__(256)
gemm_wmma_tdm(const __bf16* __restrict__ Xb, const float* __restrict__ W,
              float* __restrict__ Y) {
  __shared__ __align__(16) __bf16 As[128 * 128];
  __shared__ __align__(16) __bf16 Bs[128 * 128];

  const long row0 = (long)blockIdx.x * 128;

  if (threadIdx.x == 0) {
    const unsigned long long gaddr =
        (unsigned long long)(const void*)(Xb + row0 * 128);
    const unsigned int lds_addr = (unsigned int)(unsigned long long)(void*)As;
    // Group 0: count=1 | lds_addr | global_addr[56:0] | type=2 (bits 127:126)
    const tdm_v4u g0 = {
        1u,
        lds_addr,
        (unsigned int)(gaddr & 0xffffffffu),
        ((unsigned int)(gaddr >> 32) & 0x01ffffffu) | 0x80000000u};
    // Group 1: data_size=2B; tensor_dim0=128 (stride 128); tensor_dim1=N_NODES;
    //          tile 128x128
    const tdm_v8i g1 = {
        (int)0x00010000u,                    // data_size=1 (2 bytes)
        (int)((128u & 0xffffu) << 16),       // tensor_dim0[15:0] << 16
        0,                                   // tensor_dim0[31:16]|tensor_dim1[15:0]<<16
        (int)((N_NODES >> 16) | (128u << 16)),  // tensor_dim1[31:16] | tile_dim0<<16
        128,                                 // tile_dim1 (tile_dim2=0)
        128,                                 // tensor_dim0_stride[31:0]
        0, 0};
    const tdm_v4i z4 = {0, 0, 0, 0};
#if defined(__clang_major__) && (__clang_major__ >= 23)
    const tdm_v8i z8 = {0, 0, 0, 0, 0, 0, 0, 0};
    __builtin_amdgcn_tensor_load_to_lds(g0, g1, z4, z4, z8, 0);
#else
    __builtin_amdgcn_tensor_load_to_lds(g0, g1, z4, z4, 0);
#endif
    __builtin_amdgcn_s_wait_tensorcnt(0);
  }

  stage_W<128>(W, Bs);
  __syncthreads();

  wmma_core<128, false, false, false>(As, Bs, nullptr, Y, nullptr, row0);
}

// ---------------------------------------------------------------------------
// Self-loop contribution: AGG = HW * dinv(node)^2
// ---------------------------------------------------------------------------
__global__ void __launch_bounds__(256)
self_loop(const float* __restrict__ hw, const float* __restrict__ dinv,
          float* __restrict__ agg) {
  const long i = (long)blockIdx.x * 256 + threadIdx.x;  // over N_NODES*128
  const float di = dinv[i >> 7];
  agg[i] = hw[i] * di * di;
}

// ---------------------------------------------------------------------------
// Edge scatter: one wave per edge, 32 lanes x float4 = 128 features.
// agg[dst] += hw[src] * dinv[src]*dinv[dst]   (native f32 global atomics)
// ---------------------------------------------------------------------------
__global__ void __launch_bounds__(256)
edge_agg(const float* __restrict__ hw, const int* __restrict__ src,
         const int* __restrict__ dst, const float* __restrict__ dinv,
         float* __restrict__ agg) {
  const int lane = threadIdx.x & 31;
  const int e = (blockIdx.x * 256 + threadIdx.x) >> 5;
  if (e >= N_EDGES) return;
  const int s = src[e];
  const int d = dst[e];
  const float nrm = dinv[s] * dinv[d];
  const float4 v = *(const float4*)(hw + (long)s * 128 + lane * 4);
  float* o = agg + (long)d * 128 + lane * 4;
  unsafeAtomicAdd(o + 0, v.x * nrm);
  unsafeAtomicAdd(o + 1, v.y * nrm);
  unsafeAtomicAdd(o + 2, v.z * nrm);
  unsafeAtomicAdd(o + 3, v.w * nrm);
}

// ---------------------------------------------------------------------------
// H = relu(AGG + b); also emit the bf16 feature copy for the next TDM GEMM.
// ---------------------------------------------------------------------------
__global__ void __launch_bounds__(256)
bias_relu(const float* __restrict__ agg, const float* __restrict__ b,
          float* __restrict__ h, __bf16* __restrict__ hb) {
  const long i = (long)blockIdx.x * 256 + threadIdx.x;
  const float val = fmaxf(agg[i] + b[i & 127], 0.0f);
  h[i] = val;
  hb[i] = (__bf16)val;
}

// ---------------------------------------------------------------------------
// TopK(256/512) pool per graph: score, in-LDS bitonic sort of 512,
// pooled = mean over top-256 of h * tanh(score).
// ---------------------------------------------------------------------------
__global__ void __launch_bounds__(512)
topk_pool(const float* __restrict__ H, const float* __restrict__ p,
          float* __restrict__ pooled) {
  __shared__ float sp[128];
  __shared__ float sc[512];
  __shared__ int   si[512];
  __shared__ float spn;

  const int tid = threadIdx.x;
  const int g = blockIdx.x;

  if (tid < 128) sp[tid] = p[tid];
  __syncthreads();
  if (tid == 0) {
    float s = 0.f;
#pragma unroll 8
    for (int i = 0; i < 128; ++i) s += sp[i] * sp[i];
    spn = rsqrtf(s);
  }
  const long node = (long)g * 512 + tid;
  const float* hr = H + node * 128;
  float dacc = 0.f;
#pragma unroll 8
  for (int i = 0; i < 128; ++i) dacc += hr[i] * sp[i];
  __syncthreads();  // spn ready
  sc[tid] = dacc * spn;
  si[tid] = tid;
  __syncthreads();

  // bitonic sort, descending
  for (int k = 2; k <= 512; k <<= 1) {
    for (int j = k >> 1; j > 0; j >>= 1) {
      const int ixj = tid ^ j;
      if (ixj > tid) {
        const bool desc = ((tid & k) == 0);
        const float a = sc[tid], b = sc[ixj];
        const bool sw = desc ? (a < b) : (a > b);
        if (sw) {
          sc[tid] = b; sc[ixj] = a;
          const int t = si[tid]; si[tid] = si[ixj]; si[ixj] = t;
        }
      }
      __syncthreads();
    }
  }

  if (tid < 256) sc[tid] = tanhf(sc[tid]);
  __syncthreads();

  if (tid < 128) {
    const float* hbase = H + (long)g * 512 * 128;
    float acc = 0.f;
    for (int j = 0; j < 256; ++j) acc += hbase[(long)si[j] * 128 + tid] * sc[j];
    pooled[g * 128 + tid] = acc * (1.0f / 256.0f);
  }
}

// ---------------------------------------------------------------------------
// out[128,10] = z2[128,64] @ fc3_W[64,10] + b
// ---------------------------------------------------------------------------
__global__ void __launch_bounds__(32)
fc3_kernel(const float* __restrict__ z, const float* __restrict__ W,
           const float* __restrict__ b, float* __restrict__ out) {
  const int g = blockIdx.x;
  const int t = threadIdx.x;
  if (t < 10) {
    float acc = b[t];
#pragma unroll 8
    for (int k = 0; k < 64; ++k) acc += z[g * 64 + k] * W[k * 10 + t];
    out[g * 10 + t] = acc;
  }
}

// ---------------------------------------------------------------------------
// Host launcher
// ---------------------------------------------------------------------------
extern "C" void kernel_launch(void* const* d_in, const int* in_sizes, int n_in,
                              void* d_out, int out_size, void* d_ws, size_t ws_size,
                              hipStream_t stream) {
  const float* x      = (const float*)d_in[0];
  const int*   eidx   = (const int*)d_in[1];
  // d_in[2] = batch (unused: graphs are uniform 512-node blocks)
  const float* emb_W  = (const float*)d_in[3];
  const float* emb_b  = (const float*)d_in[4];
  const float* gcn_W  = (const float*)d_in[5];
  const float* gcn_b  = (const float*)d_in[6];
  const float* pool_p = (const float*)d_in[7];
  const float* fc1_W  = (const float*)d_in[8];
  const float* fc1_b  = (const float*)d_in[9];
  const float* fc2_W  = (const float*)d_in[10];
  const float* fc2_b  = (const float*)d_in[11];
  const float* fc3_W  = (const float*)d_in[12];
  const float* fc3_b  = (const float*)d_in[13];
  float* out = (float*)d_out;

  // Workspace layout (feature buffers stay resident in the 192MB L2):
  char* ws = (char*)d_ws;
  const size_t FEAT_BYTES = (size_t)N_NODES * HID * sizeof(float);  // 32 MB
  float*  H      = (float*)(ws);
  float*  HW     = (float*)(ws + FEAT_BYTES);
  float*  AGG    = (float*)(ws + 2 * FEAT_BYTES);
  float*  deg    = (float*)(ws + 3 * FEAT_BYTES);  // becomes dinv in place
  float*  pooled = (float*)(ws + 3 * FEAT_BYTES + (size_t)N_NODES * sizeof(float));
  float*  z1     = pooled + 128 * 128;
  float*  z2     = z1 + 128 * 128;
  __bf16* Hb     = (__bf16*)(ws + 3 * FEAT_BYTES + (1u << 20));  // 16 MB bf16 copy

  const int* src = eidx;            // edge_index[0]
  const int* dst = eidx + N_EDGES;  // edge_index[1]

  // 1) symmetric GCN normalization: dinv = rsqrt(1 + in-degree)
  deg_init   <<<N_NODES / 256, 256, 0, stream>>>(deg);
  deg_accum  <<<N_EDGES / 256, 256, 0, stream>>>(dst, deg);
  dinv_finish<<<N_NODES / 256, 256, 0, stream>>>(deg);

  // 2) embedding: H = x @ emb_W + emb_b (also produce bf16 copy for TDM GEMM)
  gemm_wmma_f32<128, true, false, true>
      <<<N_NODES / 128, 256, 0, stream>>>(x, emb_W, emb_b, H, Hb);

  // 3) GCN layers: HW = Hb @ W (A tile staged by the Tensor Data Mover)
  for (int l = 0; l < 3; ++l) {
    gemm_wmma_tdm<<<N_NODES / 128, 256, 0, stream>>>(
        Hb, gcn_W + (size_t)l * 128 * 128, HW);
    self_loop<<<(N_NODES * HID) / 256, 256, 0, stream>>>(HW, deg, AGG);
    edge_agg <<<(N_EDGES * 32) / 256, 256, 0, stream>>>(HW, src, dst, deg, AGG);
    bias_relu<<<(N_NODES * HID) / 256, 256, 0, stream>>>(AGG, gcn_b + l * 128, H, Hb);
  }

  // 4) top-k pool (ratio 0.5) + mean
  topk_pool<<<128, 512, 0, stream>>>(H, pool_p, pooled);

  // 5) MLP head
  gemm_wmma_f32<128, true, true, false><<<1, 256, 0, stream>>>(pooled, fc1_W, fc1_b, z1, nullptr);
  gemm_wmma_f32<64,  true, true, false><<<1, 256, 0, stream>>>(z1, fc2_W, fc2_b, z2, nullptr);
  fc3_kernel<<<128, 32, 0, stream>>>(z2, fc3_W, fc3_b, out);
}
